// MaskEmbdMultiMPN_Transformer_Large_88115549044887
// MI455X (gfx1250) — matrix operand order repaired and checked by reading, not compiled, not run on hardware
//
#include <hip/hip_runtime.h>
#include <math.h>

// ---------------------------------------------------------------------------
// MaskEmbdMultiMPN_Transformer_Large forward for MI455X (gfx1250, wave32).
// All large GEMMs run on v_wmma_f32_16x16x32_f16 (f16 in, fp32 accum).
// The NNConv edge-MLP is fused: per-edge (64x64) weight matrices are never
// materialized (would be 335 MB > L2); instead
//   msg = [outer(h_src,t) | h_src] @ [W2' ; B2^T]
// as a 16x4160x64 WMMA GEMM per 16-edge tile, scatter-added with f32 atomics.
// All hot-path fragment loads are contiguous 16-float runs -> global/ds b128.
// Tiny 64x64 operands needing column access (root, edge2-bias) are
// pre-transposed once per layer into workspace.
// ---------------------------------------------------------------------------

typedef __attribute__((ext_vector_type(16))) _Float16 v16h;
typedef __attribute__((ext_vector_type(8)))  float    v8f;

#define N_NODES 8192
#define HID     64
#define EDIR    10240
#define BATCH   16
#define SEQ     512
#define NHEAD   8
#define DH      8

enum { GF_GELU = 1, GF_ACC = 2 };

__device__ __forceinline__ float gelu_exact(float x) {
  return 0.5f * x * (1.0f + erff(x * 0.70710678118654752f));
}

// pack 8 consecutive f32 (two float4) into v16h[base..base+7]
__device__ __forceinline__ void pack8(v16h& a, int base, const float4& x,
                                      const float4& y) {
  a[base + 0] = (_Float16)x.x; a[base + 1] = (_Float16)x.y;
  a[base + 2] = (_Float16)x.z; a[base + 3] = (_Float16)x.w;
  a[base + 4] = (_Float16)y.x; a[base + 5] = (_Float16)y.y;
  a[base + 6] = (_Float16)y.z; a[base + 7] = (_Float16)y.w;
}
// same but scaled by s (outer-product A synthesis)
__device__ __forceinline__ void pack8s(v16h& a, int base, float s,
                                       const float4& x, const float4& y) {
  a[base + 0] = (_Float16)(s * x.x); a[base + 1] = (_Float16)(s * x.y);
  a[base + 2] = (_Float16)(s * x.z); a[base + 3] = (_Float16)(s * x.w);
  a[base + 4] = (_Float16)(s * y.x); a[base + 5] = (_Float16)(s * y.y);
  a[base + 6] = (_Float16)(s * y.z); a[base + 7] = (_Float16)(s * y.w);
}

// ---------------------------------------------------------------------------
// Generic WMMA GEMM:  Y[N x M] = op( X[N x K] @ W^T + bias ),  W (M,K) row-major.
//   GF_GELU: exact-erf GELU.  GF_ACC: Y += result.
// GUARD=false requires K%32==0 and M%16==0 -> float4 (b128) fragment loads.
// One wave per 16x16 output tile; A/B use contiguous k = 16*half + e (wave32).
// ---------------------------------------------------------------------------
template <int FLAGS, bool GUARD>
__global__ void gemm16(const float* __restrict__ X, const float* __restrict__ W,
                       const float* __restrict__ bias, float* __restrict__ Y,
                       int Nrows, int K, int M) {
  const int lane = threadIdx.x;
  const int half = lane >> 4;      // K-half (A/B) and row-half (C)
  const int l16  = lane & 15;
  const int row0 = blockIdx.x * 16;
  const int col0 = blockIdx.y * 16;
  const int arow = row0 + l16;     // A-matrix row for this lane
  const int bcol = col0 + l16;     // output column == W row

  v8f c = {};
  for (int kk = 0; kk < K; kk += 32) {
    v16h a, b;
    if (!GUARD) {
      const float* xr = X + (size_t)arow * K + kk;
      const float4* ap0 = (const float4*)(xr + half * 16);
      const float4* ap1 = (const float4*)(xr + half * 16 + 8);
      pack8(a, 0, ap0[0], ap0[1]);
      pack8(a, 8, ap1[0], ap1[1]);
      const float4* bp = (const float4*)(W + (size_t)bcol * K + kk + half * 16);
      pack8(b, 0, bp[0], bp[1]);
      pack8(b, 8, bp[2], bp[3]);
    } else {
#pragma unroll
      for (int e = 0; e < 16; ++e) {
        const int fl = half * 16 + e;
        const int k = kk + fl;
        a[e] = (_Float16)((k < K) ? X[(size_t)arow * K + k] : 0.0f);
      }
#pragma unroll
      for (int e = 0; e < 16; ++e) {
        const int k = kk + half * 16 + e;
        b[e] = (_Float16)((k < K && bcol < M) ? W[(size_t)bcol * K + k] : 0.0f);
      }
    }
    c = __builtin_amdgcn_wmma_f32_16x16x32_f16(false, a, false, b, (short)0, c,
                                               false, false);
  }

  const int col = col0 + l16;
  if (col < M) {
    const float bv = bias ? bias[col] : 0.0f;
#pragma unroll
    for (int r = 0; r < 8; ++r) {
      const int row = row0 + (half ? 8 + r : r);
      float v = c[r] + bv;
      if (FLAGS & GF_GELU) v = gelu_exact(v);
      if (FLAGS & GF_ACC) v += Y[(size_t)row * M + col];
      Y[(size_t)row * M + col] = v;
    }
  }
}

// 64x64 transpose: dst[o*64+i] = src[i*64+o]  (per-layer prep for root & b2)
__global__ void transpose64(const float* __restrict__ src, float* __restrict__ dst) {
  const int idx = blockIdx.x * blockDim.x + threadIdx.x;  // 4096 threads
  const int i = idx >> 6, o = idx & 63;
  dst[o * 64 + i] = src[i * 64 + o];
}

// ---------------------------------------------------------------------------
// Fused NNConv message kernel. One wave per tile of 16 mirrored edges.
//   t[e] = gelu(edge1(ea[e]))   (precomputed, E_DIR x 64; mirrored reuse)
//   msg[e,o] = sum_{i,k} h[src,i]*t[e,k]*W2[i*64+o,k] + sum_i h[src,i]*b2t[o*64+i]
// A-matrix (rank-1 outer product) is synthesized in registers from LDS;
// K = 128*32 (P part) + 2*32 (bias part) -> 130 WMMAs per o-tile, x4 tiles.
// Results scatter-added to agg[dst] with global f32 atomics.
// ---------------------------------------------------------------------------
__global__ void nnconv_msg(const float* __restrict__ h, const float* __restrict__ T,
                           const int* __restrict__ ei, const float* __restrict__ W2,
                           const float* __restrict__ b2t, float* __restrict__ agg,
                           int Edir) {
  __shared__ __align__(16) float hs[16 * 64];
  __shared__ __align__(16) float ts[16 * 64];
  __shared__ int dn[16];

  const int lane = threadIdx.x;
  const int e0 = blockIdx.x * 16;

  for (int i = lane; i < 16 * 64; i += 32) {
    const int r = i >> 6, cidx = i & 63;
    const int e = e0 + r;
    const int eo = (e < Edir) ? e : e - Edir;
    const int sN = (e < Edir) ? ei[eo] : ei[Edir + eo];  // mirrored src
    hs[i] = h[(size_t)sN * 64 + cidx];
    ts[i] = T[(size_t)eo * 64 + cidx];
  }
  if (lane < 16) {
    const int e = e0 + lane;
    const int eo = (e < Edir) ? e : e - Edir;
    dn[lane] = (e < Edir) ? ei[Edir + eo] : ei[eo];      // mirrored dst
  }
  __syncthreads();

  const int half = lane >> 4, l16 = lane & 15;
  for (int ot = 0; ot < 4; ++ot) {
    const int o = ot * 16 + l16;
    v8f c = {};
#pragma unroll 2
    for (int kk = 0; kk < 128; ++kk) {
      const int ib = kk >> 1;           // i index (constant within 32-chunk)
      const int kb = (kk & 1) * 32;     // k base within i-block
      const float hv = hs[l16 * 64 + ib];
      v16h a, b;
      const float4* tp0 = (const float4*)(ts + l16 * 64 + kb + half * 16);
      const float4* tp1 = (const float4*)(ts + l16 * 64 + kb + half * 16 + 8);
      pack8s(a, 0, hv, tp0[0], tp0[1]);
      pack8s(a, 8, hv, tp1[0], tp1[1]);
      const float4* wr =
          (const float4*)(W2 + ((size_t)(ib * 64 + o)) * 64 + kb + half * 16);
      pack8(b, 0, wr[0], wr[1]);
      pack8(b, 8, wr[2], wr[3]);
      c = __builtin_amdgcn_wmma_f32_16x16x32_f16(false, a, false, b, (short)0,
                                                 c, false, false);
    }
    // bias part: msg += h_src @ B2,  B2^T pre-transposed -> contiguous b128
#pragma unroll
    for (int s = 0; s < 2; ++s) {
      v16h a, b;
      const float4* hp0 = (const float4*)(hs + l16 * 64 + s * 32 + half * 16);
      const float4* hp1 = (const float4*)(hs + l16 * 64 + s * 32 + half * 16 + 8);
      pack8(a, 0, hp0[0], hp0[1]);
      pack8(a, 8, hp1[0], hp1[1]);
      const float4* bp = (const float4*)(b2t + o * 64 + s * 32 + half * 16);
      pack8(b, 0, bp[0], bp[1]);
      pack8(b, 8, bp[2], bp[3]);
      c = __builtin_amdgcn_wmma_f32_16x16x32_f16(false, a, false, b, (short)0,
                                                 c, false, false);
    }
#pragma unroll
    for (int r = 0; r < 8; ++r) {
      const int row = half ? 8 + r : r;
      atomicAdd(&agg[(size_t)dn[row] * 64 + o], c[r]);
    }
  }
}

// ---------------------------------------------------------------------------
// Small fp32 elementwise / reduction kernels (H=64 rows).
// ---------------------------------------------------------------------------
__global__ void ln_kernel(const float* __restrict__ X, const float* __restrict__ g,
                          const float* __restrict__ b, float* __restrict__ Y, int N) {
  const int row = blockIdx.x * blockDim.x + threadIdx.x;
  if (row >= N) return;
  const float* x = X + (size_t)row * 64;
  float m = 0.f;
  for (int i = 0; i < 64; ++i) m += x[i];
  m *= (1.0f / 64.0f);
  float var = 0.f;
  for (int i = 0; i < 64; ++i) { const float d = x[i] - m; var += d * d; }
  var *= (1.0f / 64.0f);
  const float inv = rsqrtf(var + 1e-5f);
  float* y = Y + (size_t)row * 64;
  for (int i = 0; i < 64; ++i) y[i] = (x[i] - m) * inv * g[i] + b[i];
}

// h = gelu(ln(A)) + h   (conv layer epilogue)
__global__ void ln_gelu_res(const float* __restrict__ A, const float* __restrict__ g,
                            const float* __restrict__ b, float* __restrict__ h, int N) {
  const int row = blockIdx.x * blockDim.x + threadIdx.x;
  if (row >= N) return;
  const float* x = A + (size_t)row * 64;
  float m = 0.f;
  for (int i = 0; i < 64; ++i) m += x[i];
  m *= (1.0f / 64.0f);
  float var = 0.f;
  for (int i = 0; i < 64; ++i) { const float d = x[i] - m; var += d * d; }
  var *= (1.0f / 64.0f);
  const float inv = rsqrtf(var + 1e-5f);
  float* y = h + (size_t)row * 64;
  for (int i = 0; i < 64; ++i)
    y[i] += gelu_exact((x[i] - m) * inv * g[i] + b[i]);
}

__global__ void add_kernel(float* __restrict__ a, const float* __restrict__ b, int n) {
  const int i = blockIdx.x * blockDim.x + threadIdx.x;
  if (i < n) a[i] += b[i];
}

__global__ void zero_kernel(float* __restrict__ a, int n) {
  const int i = blockIdx.x * blockDim.x + threadIdx.x;
  if (i < n) a[i] = 0.0f;
}

// ---------------------------------------------------------------------------
// Attention: DH=8 per head -> online-softmax VALU kernel, K/V staged in LDS.
// One workgroup per (batch, head); 128 threads, 4 query rows each.
// (0.5 GFLOP/layer total -- not worth WMMA contortions at K=8.)
// ---------------------------------------------------------------------------
__global__ void attn_kernel(const float* __restrict__ qkv, float* __restrict__ O) {
  __shared__ float Ks[SEQ * DH];
  __shared__ float Vs[SEQ * DH];
  const int bh = blockIdx.x;
  const int bi = bh >> 3, hi = bh & 7;
  const int tid = threadIdx.x;
  const float* base = qkv + (size_t)bi * SEQ * 192;

  for (int i = tid; i < SEQ * DH; i += 128) {
    const int s = i >> 3, d = i & 7;
    Ks[i] = base[(size_t)s * 192 + 64 + hi * 8 + d];
    Vs[i] = base[(size_t)s * 192 + 128 + hi * 8 + d];
  }
  __syncthreads();

  for (int q = tid; q < SEQ; q += 128) {
    float qv[DH];
#pragma unroll
    for (int d = 0; d < DH; ++d) qv[d] = base[(size_t)q * 192 + hi * 8 + d];
    float mx = -1e30f, l = 0.f, acc[DH] = {0};
    for (int j = 0; j < SEQ; ++j) {
      float s = 0.f;
#pragma unroll
      for (int d = 0; d < DH; ++d) s += qv[d] * Ks[j * 8 + d];
      s *= 0.35355339059327373f;  // 1/sqrt(8)
      const float nm = fmaxf(mx, s);
      const float sc = __expf(mx - nm);
      const float p = __expf(s - nm);
      l = l * sc + p;
#pragma unroll
      for (int d = 0; d < DH; ++d) acc[d] = acc[d] * sc + p * Vs[j * 8 + d];
      mx = nm;
    }
    const float inv = 1.0f / l;
    float* o = O + ((size_t)bi * SEQ + q) * 64 + hi * 8;
#pragma unroll
    for (int d = 0; d < DH; ++d) o[d] = acc[d] * inv;
  }
}

// ---------------------------------------------------------------------------
// Host orchestration.
// Flattened input order (depth-first dict order of setup_inputs()):
//   0 x, 1 pred_mask, 2 edge_index, 3 edge_attr, 4 batch,
//   5/6 input_encoder W/b, 7/8 mask1 W/b, 9/10 mask2 W/b,
//   11+8i pre[i]:  e1W,e1b,e2W,e2b,root,bias,ln_g,ln_b
//   35+12i tl[i]:  in_W,in_b,outW,outb,ff1W,ff1b,ff2W,ff2b,ln1g,ln1b,ln2g,ln2b
//   83+8i post[i], 99 dec_ln_g, 100 dec_ln_b, 101/102 dec1, 103/104 dec2
// ---------------------------------------------------------------------------
extern "C" void kernel_launch(void* const* d_in, const int* in_sizes, int n_in,
                              void* d_out, int out_size, void* d_ws, size_t ws_size,
                              hipStream_t stream) {
  (void)in_sizes; (void)n_in; (void)out_size; (void)ws_size;
  const int N = N_NODES;
  const float* x  = (const float*)d_in[0];
  const float* pm = (const float*)d_in[1];
  const int*   ei = (const int*)d_in[2];
  const float* ea = (const float*)d_in[3];
  auto P = [&](int i) { return (const float*)d_in[i]; };

  // workspace carve (fp32)
  float* h     = (float*)d_ws;            // N x 64
  float* u     = h     + (size_t)N * 64;  // N x 64
  float* qkv   = u     + (size_t)N * 64;  // N x 192
  float* aO    = qkv   + (size_t)N * 192; // N x 64
  float* ff    = aO    + (size_t)N * 64;  // N x 256
  float* hd    = ff    + (size_t)N * 256; // N x 64
  float* agg   = hd    + (size_t)N * 64;  // N x 64
  float* T     = agg   + (size_t)N * 64;  // EDIR x 64
  float* rootT = T     + (size_t)EDIR * 64; // 64 x 64
  float* b2t   = rootT + 64 * 64;           // 64 x 64

  const dim3 wv(32);
  auto g16 = [](int n, int m) { return dim3(n / 16, (m + 15) / 16); };
  const dim3 rblk(256);
  const dim3 rgrid((N + 255) / 256);

  // ---- encoder: h = x@Wie^T+b + mask2(gelu(mask1(pm))) ----
  gemm16<GF_GELU, true><<<g16(N, 64), wv, 0, stream>>>(pm, P(7), P(8), u, N, 6, 64);
  gemm16<0, true>      <<<g16(N, 64), wv, 0, stream>>>(x,  P(5), P(6), h, N, 6, 64);
  gemm16<GF_ACC, false><<<g16(N, 64), wv, 0, stream>>>(u,  P(9), P(10), h, N, 64, 64);

  auto conv = [&](int base) {
    // per-layer prep: transpose root (for h@root fast path) and edge2 bias
    transpose64<<<dim3(16), rblk, 0, stream>>>(P(base + 4), rootT);
    transpose64<<<dim3(16), rblk, 0, stream>>>(P(base + 3), b2t);
    gemm16<GF_GELU, true><<<g16(EDIR, 64), wv, 0, stream>>>(
        ea, P(base + 0), P(base + 1), T, EDIR, 5, 64);
    zero_kernel<<<dim3((N * 64 + 255) / 256), rblk, 0, stream>>>(agg, N * 64);
    nnconv_msg<<<dim3((2 * EDIR) / 16), wv, 0, stream>>>(h, T, ei, P(base + 2),
                                                         b2t, agg, EDIR);
    gemm16<GF_ACC, false><<<g16(N, 64), wv, 0, stream>>>(h, rootT, P(base + 5),
                                                         agg, N, 64, 64);
    ln_gelu_res<<<rgrid, rblk, 0, stream>>>(agg, P(base + 6), P(base + 7), h, N);
  };
  conv(11); conv(19); conv(27);   // pre convs

  hipMemcpyAsync(hd, h, (size_t)N * 64 * sizeof(float),
                 hipMemcpyDeviceToDevice, stream);

  auto tl = [&](int base) {
    ln_kernel<<<rgrid, rblk, 0, stream>>>(hd, P(base + 8), P(base + 9), u, N);
    gemm16<0, false><<<g16(N, 192), wv, 0, stream>>>(u, P(base + 0), P(base + 1),
                                                     qkv, N, 64, 192);
    attn_kernel<<<dim3(BATCH * NHEAD), dim3(128), 0, stream>>>(qkv, aO);
    gemm16<GF_ACC, false><<<g16(N, 64), wv, 0, stream>>>(aO, P(base + 2),
                                                         P(base + 3), hd, N, 64, 64);
    ln_kernel<<<rgrid, rblk, 0, stream>>>(hd, P(base + 10), P(base + 11), u, N);
    gemm16<GF_GELU, false><<<g16(N, 256), wv, 0, stream>>>(u, P(base + 4),
                                                           P(base + 5), ff, N, 64, 256);
    gemm16<GF_ACC, false><<<g16(N, 64), wv, 0, stream>>>(ff, P(base + 6),
                                                         P(base + 7), hd, N, 256, 64);
  };
  tl(35); tl(47); tl(59); tl(71);

  add_kernel<<<dim3((N * 64 + 255) / 256), rblk, 0, stream>>>(h, hd, N * 64);

  conv(83); conv(91);             // post convs

  // ---- decoder ----
  ln_kernel<<<rgrid, rblk, 0, stream>>>(h, P(99), P(100), u, N);
  gemm16<GF_GELU, false><<<g16(N, 64), wv, 0, stream>>>(u, P(101), P(102), ff,
                                                        N, 64, 64);
  gemm16<0, true><<<g16(N, 6), wv, 0, stream>>>(ff, P(103), P(104),
                                                (float*)d_out, N, 64, 6);
}
